// Attention_43430709297174
// MI455X (gfx1250) — matrix operand order repaired and checked by reading, not compiled, not run on hardware
//
#include <hip/hip_runtime.h>
#include <hip/hip_bf16.h>

// ---------------------------------------------------------------------------
// SAGAN self-attention block for MI455X (gfx1250, wave32, WMMA).
// Pipeline:
//   k1: spectral-norm scalars (4 tiny power-iteration norms -> 1/sigma)
//   k2: conv1x1 f32-WMMA (theta/phi/g), outputs scaled by 1/sigma, bf16
//   k3: 2x2 maxpool (phi, g)
//   k4: fused flash attention: S=theta^T phi (bf16 WMMA), online softmax,
//       O += P g^T (bf16 WMMA)  -- beta (134MB) never hits HBM
//   k5: conv1x1 f32-WMMA with w_o + residual epilogue gamma*o + x
// ---------------------------------------------------------------------------

typedef float v8f  __attribute__((ext_vector_type(8)));
typedef float v2f  __attribute__((ext_vector_type(2)));
typedef __bf16 v16bf __attribute__((ext_vector_type(16)));
typedef unsigned short us16v __attribute__((ext_vector_type(16)));

__device__ __forceinline__ unsigned short f32_to_bf16(float f) {
  unsigned u = __float_as_uint(f);
  u += 0x7FFFu + ((u >> 16) & 1u);          // round-to-nearest-even
  return (unsigned short)(u >> 16);
}
__device__ __forceinline__ float bf16_to_f32(unsigned short h) {
  return __uint_as_float(((unsigned)h) << 16);
}

// ---------------------------------------------------------------------------
// Kernel 1: spectral norms.  sigma = || W * (uW / ||uW||) ||  (one iter, eval)
// One 256-thread block per weight matrix.
// ---------------------------------------------------------------------------
__global__ __launch_bounds__(256) void sn_kernel(
    const float* __restrict__ w_theta, const float* __restrict__ w_phi,
    const float* __restrict__ w_g,     const float* __restrict__ w_o,
    const float* __restrict__ u_theta, const float* __restrict__ u_phi,
    const float* __restrict__ u_g,     const float* __restrict__ u_o,
    float* __restrict__ inv_sv) {
  __shared__ float v[512];
  __shared__ float red[256];
  const float* Ws[4] = {w_theta, w_phi, w_g, w_o};
  const float* Us[4] = {u_theta, u_phi, u_g, u_o};
  const int Os[4] = {64, 64, 256, 512};
  const int Is[4] = {512, 512, 512, 256};
  const int which = blockIdx.x;
  const float* W = Ws[which];
  const float* U = Us[which];
  const int O = Os[which], I = Is[which];
  const int t = threadIdx.x;

  // v = u @ W  (length I)
  for (int j = t; j < I; j += 256) {
    float s = 0.f;
    for (int o = 0; o < O; ++o) s += U[o] * W[(size_t)o * I + j];
    v[j] = s;
  }
  __syncthreads();
  float part = 0.f;
  for (int j = t; j < I; j += 256) part += v[j] * v[j];
  red[t] = part;
  __syncthreads();
  for (int s = 128; s > 0; s >>= 1) { if (t < s) red[t] += red[t + s]; __syncthreads(); }
  const float inv_vn = 1.f / fmaxf(sqrtf(red[0]), 1e-12f);
  __syncthreads();

  // tvec = W @ v_hat ; sigma = tvec . (tvec/max(||tvec||,eps))
  float part2 = 0.f;
  for (int o = t; o < O; o += 256) {
    float s = 0.f;
    for (int j = 0; j < I; ++j) s += W[(size_t)o * I + j] * v[j];
    s *= inv_vn;
    part2 += s * s;
  }
  red[t] = part2;
  __syncthreads();
  for (int s = 128; s > 0; s >>= 1) { if (t < s) red[t] += red[t + s]; __syncthreads(); }
  if (t == 0) {
    float tn2 = red[0];
    float sv = tn2 / fmaxf(sqrtf(tn2), 1e-12f);
    inv_sv[which] = 1.f / sv;
  }
}

// ---------------------------------------------------------------------------
// Kernel 2: conv1x1 as f32 WMMA GEMM:  Out[b][O][N] = scale * (W[O,K] X_b[K,N])
// One wave per block; wave computes a 16(O) x 64(N) tile.  Output bf16.
// f32 16x16x4 fragments: A/B element i lives at K = i + 2*half.
// ---------------------------------------------------------------------------
__global__ __launch_bounds__(32) void conv1x1_f32_bf16out(
    const float* __restrict__ W, const float* __restrict__ X,
    unsigned short* __restrict__ Out, const float* __restrict__ scale_p,
    int O, int K) {
  const int N = 4096;
  const int lane = threadIdx.x & 31;
  const int l15 = lane & 15, half = lane >> 4;
  const int n0 = blockIdx.x * 64;
  const int o0 = blockIdx.y * 16;
  const int b  = blockIdx.z;
  const float* Xb = X + (size_t)b * K * N;
  const float scale = scale_p[0];

  v8f acc[4] = {};
  for (int k0 = 0; k0 < K; k0 += 4) {
    v2f a;
    const size_t wrow = (size_t)(o0 + l15) * K + k0 + 2 * half;
    a.x = W[wrow + 0];
    a.y = W[wrow + 1];
#pragma unroll
    for (int tt = 0; tt < 4; ++tt) {
      v2f bb;
      bb.x = Xb[(size_t)(k0 + 2 * half + 0) * N + n0 + tt * 16 + l15];
      bb.y = Xb[(size_t)(k0 + 2 * half + 1) * N + n0 + tt * 16 + l15];
      acc[tt] = __builtin_amdgcn_wmma_f32_16x16x4_f32(
          false, a, false, bb, (short)0, acc[tt], false, false);
    }
  }
#pragma unroll
  for (int tt = 0; tt < 4; ++tt)
#pragma unroll
    for (int i = 0; i < 8; ++i) {
      float valf = acc[tt][i] * scale;           // D row = i + 8*half, col = l15
      Out[((size_t)b * O + o0 + i + 8 * half) * N + n0 + tt * 16 + l15] =
          f32_to_bf16(valf);
    }
}

// ---------------------------------------------------------------------------
// Kernel 3: 2x2 maxpool on bf16, [BC][64][64] -> [BC][32][32]
// ---------------------------------------------------------------------------
__global__ void pool2_bf16(const unsigned short* __restrict__ In,
                           unsigned short* __restrict__ Out, int total) {
  int idx = blockIdx.x * blockDim.x + threadIdx.x;
  if (idx >= total) return;
  int wp = idx & 31, hp = (idx >> 5) & 31, bc = idx >> 10;
  const unsigned short* p = In + (size_t)bc * 4096 + (size_t)hp * 128 + wp * 2;
  unsigned short best = p[0];
  float m = bf16_to_f32(best);
  unsigned short c1 = p[1], c2 = p[64], c3 = p[65];
  float f1 = bf16_to_f32(c1), f2 = bf16_to_f32(c2), f3 = bf16_to_f32(c3);
  if (f1 > m) { m = f1; best = c1; }
  if (f2 > m) { m = f2; best = c2; }
  if (f3 > m) { m = f3; best = c3; }
  Out[idx] = best;
}

// ---------------------------------------------------------------------------
// Kernel 4: fused flash attention.
//  Th [B][64][4096] bf16 (theta, scaled), Ph [B][64][1024], G [B][256][1024]
//  Oa [B][256][4096] f32:  Oa[c,n] = sum_m softmax_m(theta^T phi)[n,m] g[c,m]
//  8 waves/block, each wave owns 16 rows (n) and the full dv=256 accumulator.
// ---------------------------------------------------------------------------
__global__ __launch_bounds__(256) void attn_kernel(
    const unsigned short* __restrict__ Th, const unsigned short* __restrict__ Ph,
    const unsigned short* __restrict__ G, float* __restrict__ Oa) {
  __shared__ unsigned short lds_p[8][16 * 32];   // per-wave P staging (D->A transpose)
  const int wave = threadIdx.x >> 5;
  const int lane = threadIdx.x & 31;
  const int l15 = lane & 15, half = lane >> 4;
  const int b = blockIdx.y;
  const int n0 = (blockIdx.x * 8 + wave) * 16;
  const unsigned short* Thb = Th + (size_t)b * 64 * 4096;
  const unsigned short* Phb = Ph + (size_t)b * 64 * 1024;
  const unsigned short* Gb  = G  + (size_t)b * 256 * 1024;

  // theta A-fragments (rows = n-pixels, K = channels 0..63), loaded once
  us16v a0u, a1u;
#pragma unroll
  for (int i = 0; i < 16; ++i) {
    int k = i + 8 * half + ((i >= 8) ? 8 : 0);   // 16-bit A 16x32 layout
    a0u[i] = Thb[(size_t)(k +  0) * 4096 + n0 + l15];
    a1u[i] = Thb[(size_t)(k + 32) * 4096 + n0 + l15];
  }
  const v16bf aTh0 = __builtin_bit_cast(v16bf, a0u);
  const v16bf aTh1 = __builtin_bit_cast(v16bf, a1u);

  v8f acc[16];
#pragma unroll
  for (int tg = 0; tg < 16; ++tg) acc[tg] = (v8f){};
  float rmax[8], rsum[8];
#pragma unroll
  for (int i = 0; i < 8; ++i) { rmax[i] = -1e30f; rsum[i] = 0.f; }

  for (int m0 = 0; m0 < 1024; m0 += 32) {
    // ---- S = theta^T phi for a 16n x 32m slab (4 bf16 WMMAs, K=64) ----
    v8f s[2];
#pragma unroll
    for (int mt = 0; mt < 2; ++mt) {
      us16v b0u, b1u;
#pragma unroll
      for (int i = 0; i < 16; ++i) {
        int kc = i + 8 * half + ((i >= 8) ? 8 : 0);  // 16-bit B 32x16 layout
        b0u[i] = Phb[(size_t)(kc +  0) * 1024 + m0 + mt * 16 + l15];
        b1u[i] = Phb[(size_t)(kc + 32) * 1024 + m0 + mt * 16 + l15];
      }
      v8f z = (v8f){};
      z = __builtin_amdgcn_wmma_f32_16x16x32_bf16(
          false, aTh0, false, __builtin_bit_cast(v16bf, b0u), (short)0, z, false, false);
      z = __builtin_amdgcn_wmma_f32_16x16x32_bf16(
          false, aTh1, false, __builtin_bit_cast(v16bf, b1u), (short)0, z, false, false);
      s[mt] = z;
    }

    // ---- online softmax: row r = i + 8*half lives across the 16 lanes of a half ----
#pragma unroll
    for (int i = 0; i < 8; ++i) {
      float mloc = fmaxf(s[0][i], s[1][i]);
      for (int d = 1; d < 16; d <<= 1) mloc = fmaxf(mloc, __shfl_xor(mloc, d, 32));
      float nm = fmaxf(rmax[i], mloc);
      float corr = __expf(rmax[i] - nm);
      float p0 = __expf(s[0][i] - nm);
      float p1 = __expf(s[1][i] - nm);
      float ps = p0 + p1;
      for (int d = 1; d < 16; d <<= 1) ps += __shfl_xor(ps, d, 32);
      rsum[i] = rsum[i] * corr + ps;
      rmax[i] = nm;
#pragma unroll
      for (int tg = 0; tg < 16; ++tg) acc[tg][i] *= corr;
      int r = i + 8 * half;
      lds_p[wave][r * 32 +  0 + l15] = f32_to_bf16(p0);
      lds_p[wave][r * 32 + 16 + l15] = f32_to_bf16(p1);
    }
    asm volatile("s_wait_dscnt 0" ::: "memory");  // P staged before re-read

    // ---- reload P as A-fragment (16n x 32m) ----
    us16v pu;
#pragma unroll
    for (int i = 0; i < 16; ++i) {
      int k = i + 16 * half;
      pu[i] = lds_p[wave][l15 * 32 + k];
    }
    const v16bf pfrag = __builtin_bit_cast(v16bf, pu);

    // ---- O += P * g^T over dv=256 (16 bf16 WMMAs, K=32) ----
#pragma unroll
    for (int tg = 0; tg < 16; ++tg) {
      us16v gu;
#pragma unroll
      for (int i = 0; i < 16; ++i)
        gu[i] = Gb[(size_t)(tg * 16 + l15) * 1024 + m0 + 16 * half + i];
      acc[tg] = __builtin_amdgcn_wmma_f32_16x16x32_bf16(
          false, pfrag, false, __builtin_bit_cast(v16bf, gu), (short)0,
          acc[tg], false, false);
    }
    if (m0 + 32 < 1024)
      __builtin_prefetch(Phb + (size_t)(16 * half) * 1024 + m0 + 32 + l15, 0, 1);
  }

  // ---- normalize and store: Oa[b][cg][n] ----
#pragma unroll
  for (int tg = 0; tg < 16; ++tg)
#pragma unroll
    for (int i = 0; i < 8; ++i) {
      float val = acc[tg][i] / rsum[i];          // D row = n-local = i+8*half
      Oa[((size_t)b * 256 + tg * 16 + l15) * 4096 + n0 + i + 8 * half] = val;
    }
}

// ---------------------------------------------------------------------------
// Kernel 5: final conv (w_o, K=256, O=512) + residual: out = gamma*o + x
// ---------------------------------------------------------------------------
__global__ __launch_bounds__(32) void conv1x1_final(
    const float* __restrict__ W, const float* __restrict__ X,   // X = o_attn
    const float* __restrict__ xin, const float* __restrict__ gamma_p,
    const float* __restrict__ scale_p, float* __restrict__ Out) {
  const int N = 4096, K = 256, O = 512;
  const int lane = threadIdx.x & 31;
  const int l15 = lane & 15, half = lane >> 4;
  const int n0 = blockIdx.x * 64;
  const int o0 = blockIdx.y * 16;
  const int b  = blockIdx.z;
  const float* Xb = X + (size_t)b * K * N;
  const float scale = scale_p[0];
  const float gamma = gamma_p[0];

  v8f acc[4] = {};
  for (int k0 = 0; k0 < K; k0 += 4) {
    v2f a;
    const size_t wrow = (size_t)(o0 + l15) * K + k0 + 2 * half;
    a.x = W[wrow + 0];
    a.y = W[wrow + 1];
#pragma unroll
    for (int tt = 0; tt < 4; ++tt) {
      v2f bb;
      bb.x = Xb[(size_t)(k0 + 2 * half + 0) * N + n0 + tt * 16 + l15];
      bb.y = Xb[(size_t)(k0 + 2 * half + 1) * N + n0 + tt * 16 + l15];
      acc[tt] = __builtin_amdgcn_wmma_f32_16x16x4_f32(
          false, a, false, bb, (short)0, acc[tt], false, false);
    }
  }
#pragma unroll
  for (int tt = 0; tt < 4; ++tt)
#pragma unroll
    for (int i = 0; i < 8; ++i) {
      size_t idx = ((size_t)b * O + o0 + i + 8 * half) * N + n0 + tt * 16 + l15;
      Out[idx] = gamma * (acc[tt][i] * scale) + xin[idx];
    }
}

// ---------------------------------------------------------------------------
extern "C" void kernel_launch(void* const* d_in, const int* in_sizes, int n_in,
                              void* d_out, int out_size, void* d_ws, size_t ws_size,
                              hipStream_t stream) {
  (void)in_sizes; (void)n_in; (void)out_size; (void)ws_size;
  const float* x       = (const float*)d_in[0];
  const float* w_theta = (const float*)d_in[1];
  const float* w_phi   = (const float*)d_in[2];
  const float* w_g     = (const float*)d_in[3];
  const float* w_o     = (const float*)d_in[4];
  const float* u_theta = (const float*)d_in[5];
  const float* u_phi   = (const float*)d_in[6];
  const float* u_g     = (const float*)d_in[7];
  const float* u_o     = (const float*)d_in[8];
  const float* gamma   = (const float*)d_in[9];
  float* out = (float*)d_out;

  char* ws = (char*)d_ws;
  float* inv_sv = (float*)ws;                                   // 4 scalars
  unsigned short* theta_bf = (unsigned short*)(ws + 256);       // [8][64][4096]   4.2MB
  unsigned short* phi_p = theta_bf + (size_t)8 * 64 * 4096;     // [8][64][1024]   1.0MB
  unsigned short* g_p   = phi_p   + (size_t)8 * 64 * 1024;      // [8][256][1024]  4.2MB
  char* big = (char*)(g_p + (size_t)8 * 256 * 1024);
  unsigned short* phi_full = (unsigned short*)big;              // [8][64][4096]  (dead after pool)
  unsigned short* g_full   = phi_full + (size_t)8 * 64 * 4096;  // [8][256][4096] (dead after pool)
  float* o_attn = (float*)big;                                  // [8][256][4096] 33.5MB (reuses big)

  sn_kernel<<<4, 256, 0, stream>>>(w_theta, w_phi, w_g, w_o,
                                   u_theta, u_phi, u_g, u_o, inv_sv);

  conv1x1_f32_bf16out<<<dim3(64, 4, 8), 32, 0, stream>>>(
      w_theta, x, theta_bf, inv_sv + 0, 64, 512);
  conv1x1_f32_bf16out<<<dim3(64, 4, 8), 32, 0, stream>>>(
      w_phi, x, phi_full, inv_sv + 1, 64, 512);
  conv1x1_f32_bf16out<<<dim3(64, 16, 8), 32, 0, stream>>>(
      w_g, x, g_full, inv_sv + 2, 256, 512);

  pool2_bf16<<<(8 * 64 * 1024 + 255) / 256, 256, 0, stream>>>(
      phi_full, phi_p, 8 * 64 * 1024);
  pool2_bf16<<<(8 * 256 * 1024 + 255) / 256, 256, 0, stream>>>(
      g_full, g_p, 8 * 256 * 1024);

  attn_kernel<<<dim3(32, 8), 256, 0, stream>>>(theta_bf, phi_p, g_p, o_attn);

  conv1x1_final<<<dim3(64, 32, 8), 32, 0, stream>>>(
      w_o, o_attn, x, gamma, inv_sv + 3, out);
}
